// SceneFun3DPN2_13503377178962
// MI455X (gfx1250) — matrix operand order repaired and checked by reading, not compiled, not run on hardware
//
#include <hip/hip_runtime.h>
#include <hip/hip_bf16.h>
#include <math.h>

typedef __attribute__((ext_vector_type(16))) _Float16 v16h;
typedef __attribute__((ext_vector_type(8)))  float    v8f;

#define BB   8
#define N0   4096
#define NCLS 25
#define GEMM_NT 4   // 16x64 output per wave (4 WMMA tiles, always all active)

// ---------------------------------------------------------------------------
// Pack W[K,N] (f32 row-major) into per-lane-contiguous f16 B fragments,
// zero-padded to Kpad = KC*32 and TNpad (multiple of GEMM_NT) tiles of 16 cols.
// Fragment address: Wp + (((tn*KC)+kc)*32 + lane)*16, j=0..15 ->
//   K = kc*32 + ((lane&16)?16:0) + j,  col = tn*16 + (lane&15)
// Out-of-range (K or col) entries are zero, so the GEMM needs no B guards.
// ---------------------------------------------------------------------------
__global__ void pack_w_kernel(const float* __restrict__ W, int K, int N,
                              _Float16* __restrict__ Wp, int KC, int TNpad) {
  int i = blockIdx.x * blockDim.x + threadIdx.x;  // (tn, kc, lane)
  int total = TNpad * KC * 32;
  if (i >= total) return;
  int lane = i & 31;
  int rest = i >> 5;
  int kc = rest % KC, tn = rest / KC;
  int colB = tn * 16 + (lane & 15);
  int kbBase = (lane & 16) ? 16 : 0;
  _Float16* o = Wp + (size_t)i * 16;
#pragma unroll
  for (int j = 0; j < 16; ++j) {
    int kb = kc * 32 + kbBase + j;
    float v = (kb < K && colB < N) ? W[(size_t)kb * N + colB] : 0.0f;
    o[j] = (_Float16)v;
  }
}

// ---------------------------------------------------------------------------
// WMMA GEMM: Y[M,N] = A[M,K] @ W[K,N]; A is f32 (converted on the fly),
// W pre-packed zero-padded f16 fragments. One wave per 16x64 output block,
// 4 accumulators, fully branch-free K loop (unrolled x2 for load/WMMA overlap).
// M must be a multiple of 16 (true at every call site).
// ---------------------------------------------------------------------------
__global__ __launch_bounds__(32) void wmma_gemm_kernel(
    const float* __restrict__ A, const _Float16* __restrict__ Wp,
    float* __restrict__ Y, int M, int K, int N, int KC, int TNG) {
  const int tile = blockIdx.x;
  const int tm = tile / TNG, tg = tile % TNG;
  const int lane = threadIdx.x & 31;
  const int rowA = tm * 16 + (lane & 15);
  const int kaBase = (lane & 16) ? 16 : 0;  // contiguous K base (matches packed B)
  const float* arow = A + (size_t)rowA * K;
  const _Float16* wbase = Wp + ((size_t)tg * GEMM_NT * KC * 32 + lane) * 16;

  v8f acc[GEMM_NT] = {};
  const int Kmain = K & ~31;
  int kc = 0;
#pragma unroll 2
  for (int k0 = 0; k0 < Kmain; k0 += 32, ++kc) {
    v16h a;
#pragma unroll
    for (int j = 0; j < 16; ++j)
      a[j] = (_Float16)arow[k0 + kaBase + j];
#pragma unroll
    for (int t = 0; t < GEMM_NT; ++t) {
      v16h b = *(const v16h*)(wbase + ((size_t)t * KC + kc) * 32 * 16);
      acc[t] = __builtin_amdgcn_wmma_f32_16x16x32_f16(
          false, a, false, b, (short)0, acc[t], false, false);
    }
  }
  if (Kmain < K) {  // single guarded tail chunk (A only; B is zero-padded)
    v16h a;
#pragma unroll
    for (int j = 0; j < 16; ++j) {
      int ka = Kmain + kaBase + j;
      a[j] = (_Float16)((ka < K) ? arow[ka] : 0.0f);
    }
#pragma unroll
    for (int t = 0; t < GEMM_NT; ++t) {
      v16h b = *(const v16h*)(wbase + ((size_t)t * KC + kc) * 32 * 16);
      acc[t] = __builtin_amdgcn_wmma_f32_16x16x32_f16(
          false, a, false, b, (short)0, acc[t], false, false);
    }
  }
  const int mBase = tm * 16 + ((lane & 16) ? 8 : 0);
#pragma unroll
  for (int t = 0; t < GEMM_NT; ++t) {
    int colB = (tg * GEMM_NT + t) * 16 + (lane & 15);
    if (colB < N) {
#pragma unroll
      for (int i = 0; i < 8; ++i)
        Y[(size_t)(mBase + i) * N + colB] = acc[t][i];
    }
  }
}

// ---------------------------------------------------------------------------
// (B,3,N) -> (B,N,3)
// ---------------------------------------------------------------------------
__global__ void xyz_transpose_kernel(const float* __restrict__ in,
                                     float* __restrict__ out) {
  int i = blockIdx.x * blockDim.x + threadIdx.x;
  if (i >= BB * N0 * 3) return;
  int b = i / (N0 * 3), r = i % (N0 * 3);
  int n = r / 3, c = r % 3;
  out[i] = in[((size_t)b * 3 + c) * N0 + n];
}

// ---------------------------------------------------------------------------
// Farthest point sampling; one block per batch, LDS distance array.
// ---------------------------------------------------------------------------
#define FPS_T 256
__global__ void fps_kernel(const float* __restrict__ xyz, int N, int npoint,
                           int* __restrict__ idx_out) {
  __shared__ float dist[4096];
  __shared__ float sval[FPS_T];
  __shared__ int   sidx[FPS_T];
  __shared__ int   s_far;
  int b = blockIdx.x, t = threadIdx.x;
  const float* X = xyz + (size_t)b * N * 3;
  for (int i = t; i < N; i += FPS_T) dist[i] = 1e10f;
  if (t == 0) s_far = 0;
  __syncthreads();
  for (int it = 0; it < npoint; ++it) {
    int far = s_far;
    if (t == 0) idx_out[b * npoint + it] = far;
    float fx = X[far * 3], fy = X[far * 3 + 1], fz = X[far * 3 + 2];
    float bestv = -1.0f; int besti = 0;
    for (int i = t; i < N; i += FPS_T) {
      float dx = X[i * 3] - fx, dy = X[i * 3 + 1] - fy, dz = X[i * 3 + 2] - fz;
      float nd = fminf(dist[i], dx * dx + dy * dy + dz * dz);
      dist[i] = nd;
      if (nd > bestv) { bestv = nd; besti = i; }
    }
    sval[t] = bestv; sidx[t] = besti;
    __syncthreads();
    for (int s = FPS_T / 2; s > 0; s >>= 1) {
      if (t < s) {
        if (sval[t + s] > sval[t] ||
            (sval[t + s] == sval[t] && sidx[t + s] < sidx[t])) {
          sval[t] = sval[t + s]; sidx[t] = sidx[t + s];
        }
      }
      __syncthreads();
    }
    if (t == 0) s_far = sidx[0];
    __syncthreads();
  }
}

__global__ void gather_xyz_kernel(const float* __restrict__ xyz,
                                  const int* __restrict__ idx,
                                  float* __restrict__ out,
                                  int Npts, int S, int total) {
  int i = blockIdx.x * blockDim.x + threadIdx.x;   // i = b*S + s
  if (i >= total) return;
  int b = i / S;
  int g = idx[i];
  const float* p = xyz + ((size_t)b * Npts + g) * 3;
  out[i * 3 + 0] = p[0]; out[i * 3 + 1] = p[1]; out[i * 3 + 2] = p[2];
}

// First Kn points (in index order) within radius; pad with first hit.
__global__ void ball_query_kernel(const float* __restrict__ xyz,
                                  const float* __restrict__ nxyz,
                                  int Npts, int S, int Kn, float r2,
                                  int* __restrict__ gidx, int total) {
  int i = blockIdx.x * blockDim.x + threadIdx.x;   // i = b*S + s
  if (i >= total) return;
  int b = i / S;
  const float* X = xyz + (size_t)b * Npts * 3;
  float cx = nxyz[i * 3], cy = nxyz[i * 3 + 1], cz = nxyz[i * 3 + 2];
  int* out = gidx + (size_t)i * Kn;
  int cnt = 0, first = 0;
  for (int n = 0; n < Npts && cnt < Kn; ++n) {
    float dx = X[n * 3] - cx, dy = X[n * 3 + 1] - cy, dz = X[n * 3 + 2] - cz;
    if (dx * dx + dy * dy + dz * dz <= r2) {
      if (cnt == 0) first = n;
      out[cnt++] = n;
    }
  }
  for (int k = cnt; k < Kn; ++k) out[k] = first;
}

// gp row = [pts[g] (Cp) | xyz[g]-center (3)]
__global__ void group_kernel(const float* __restrict__ xyz,
                             const float* __restrict__ pts, int Cp,
                             const float* __restrict__ nxyz,
                             const int* __restrict__ gidx,
                             int S, int Kn, int Npts,
                             float* __restrict__ gp, size_t total) {
  size_t i = (size_t)blockIdx.x * blockDim.x + threadIdx.x; // b*S*Kn + ...
  if (i >= total) return;
  int k = (int)(i % Kn);
  size_t bs = i / Kn;           // b*S + s
  int b = (int)(bs / S);
  int g = gidx[bs * Kn + k];
  const float* p = pts + ((size_t)b * Npts + g) * Cp;
  float* o = gp + i * (Cp + 3);
  for (int c = 0; c < Cp; ++c) o[c] = p[c];
  const float* xg = xyz + ((size_t)b * Npts + g) * 3;
  const float* cx = nxyz + bs * 3;
  o[Cp + 0] = xg[0] - cx[0];
  o[Cp + 1] = xg[1] - cx[1];
  o[Cp + 2] = xg[2] - cx[2];
}

// Per-channel mean / rstd over all M rows.
__global__ void bn_stats_kernel(const float* __restrict__ Y, int M, int C,
                                float* __restrict__ stats) {
  __shared__ float s1[256], s2[256];
  int c = blockIdx.x, t = threadIdx.x;
  float a = 0.f, b = 0.f;
  for (int m = t; m < M; m += 256) {
    float v = Y[(size_t)m * C + c];
    a += v; b += v * v;
  }
  s1[t] = a; s2[t] = b;
  __syncthreads();
  for (int s = 128; s > 0; s >>= 1) {
    if (t < s) { s1[t] += s1[t + s]; s2[t] += s2[t + s]; }
    __syncthreads();
  }
  if (t == 0) {
    float mean = s1[0] / (float)M;
    float var  = s2[0] / (float)M - mean * mean;
    stats[c] = mean;
    stats[C + c] = rsqrtf(fmaxf(var, 0.f) + 1e-5f);
  }
}

__global__ void bn_apply_kernel(float* __restrict__ Y,
                                const float* __restrict__ stats,
                                size_t total, int C, int relu) {
  size_t i = (size_t)blockIdx.x * blockDim.x + threadIdx.x;
  if (i >= total) return;
  int c = (int)(i % C);
  float v = (Y[i] - stats[c]) * stats[C + c];
  Y[i] = relu ? fmaxf(v, 0.f) : v;
}

// max over the Kn "sample" axis; writes into a wider concat buffer.
__global__ void maxpool_kernel(const float* __restrict__ H, int Kn, int C,
                               float* __restrict__ out, int outStride,
                               int colOff, int BS) {
  int i = blockIdx.x * blockDim.x + threadIdx.x;  // over BS*C
  if (i >= BS * C) return;
  int bs = i / C, c = i % C;
  const float* h = H + ((size_t)bs * Kn) * C + c;
  float m = -1e30f;
  for (int k = 0; k < Kn; ++k) m = fmaxf(m, h[(size_t)k * C]);
  out[(size_t)bs * outStride + colOff + c] = m;
}

// out row = [A row (Ca) | B row (Cb)]; bcastRows>0: B row index = r/bcastRows.
__global__ void concat2_kernel(const float* __restrict__ A, int Ca,
                               const float* __restrict__ Bp, int Cb,
                               float* __restrict__ out, int rows,
                               int bcastRows) {
  int r = blockIdx.x * blockDim.x + threadIdx.x;
  if (r >= rows) return;
  const float* a = A + (size_t)r * Ca;
  int br = bcastRows ? (r / bcastRows) : r;
  const float* bp = Bp + (size_t)br * Cb;
  float* o = out + (size_t)r * (Ca + Cb);
  for (int c = 0; c < Ca; ++c) o[c] = a[c];
  for (int c = 0; c < Cb; ++c) o[Ca + c] = bp[c];
}

// 3-NN inverse-distance interpolation + concat with skip features.
__global__ void fp_interp_kernel(const float* __restrict__ xyz1, int Nn,
                                 const float* __restrict__ xyz2, int S,
                                 const float* __restrict__ pts1, int C1,
                                 const float* __restrict__ pts2, int C2,
                                 float* __restrict__ out, int total) {
  int i = blockIdx.x * blockDim.x + threadIdx.x;  // b*Nn + n
  if (i >= total) return;
  int b = i / Nn;
  const float* q = xyz1 + (size_t)i * 3;
  float qx = q[0], qy = q[1], qz = q[2];
  const float* X2 = xyz2 + (size_t)b * S * 3;
  float d0 = 1e30f, d1 = 1e30f, d2 = 1e30f;
  int j0 = 0, j1 = 0, j2 = 0;
  for (int s = 0; s < S; ++s) {
    float dx = X2[s * 3] - qx, dy = X2[s * 3 + 1] - qy, dz = X2[s * 3 + 2] - qz;
    float d = dx * dx + dy * dy + dz * dz;
    if (d < d0)      { d2 = d1; j2 = j1; d1 = d0; j1 = j0; d0 = d; j0 = s; }
    else if (d < d1) { d2 = d1; j2 = j1; d1 = d; j1 = s; }
    else if (d < d2) { d2 = d; j2 = s; }
  }
  float w0 = 1.f / (d0 + 1e-8f), w1 = 1.f / (d1 + 1e-8f), w2 = 1.f / (d2 + 1e-8f);
  float ws = w0 + w1 + w2;
  w0 /= ws; w1 /= ws; w2 /= ws;
  const float* p0 = pts2 + ((size_t)b * S + j0) * C2;
  const float* p1 = pts2 + ((size_t)b * S + j1) * C2;
  const float* p2 = pts2 + ((size_t)b * S + j2) * C2;
  float* o = out + (size_t)i * (C1 + C2);
  const float* a = pts1 + (size_t)i * C1;
  for (int c = 0; c < C1; ++c) o[c] = a[c];
  for (int c = 0; c < C2; ++c)
    o[C1 + c] = w0 * p0[c] + w1 * p1[c] + w2 * p2[c];
}

// Column norms of text_features (512 x 25).
__global__ void tnorm_kernel(const float* __restrict__ T, float* __restrict__ tn) {
  int k = threadIdx.x;
  if (k < NCLS) {
    float s = 0.f;
    for (int j = 0; j < 512; ++j) { float v = T[j * NCLS + k]; s += v * v; }
    tn[k] = sqrtf(s);
  }
}

// Cosine-sim scaling + log_softmax + transpose to (B, K, N).
__global__ void final_kernel(const float* __restrict__ H,
                             const float* __restrict__ simY,
                             const float* __restrict__ tn,
                             const float* __restrict__ lsp,
                             float* __restrict__ out,
                             int Np, int Kc, int Ch, int total) {
  int i = blockIdx.x * blockDim.x + threadIdx.x;  // b*Np + n
  if (i >= total) return;
  int b = i / Np, n = i % Np;
  const float* h = H + (size_t)i * Ch;
  float s = 0.f;
  for (int c = 0; c < Ch; ++c) s += h[c] * h[c];
  float hn = sqrtf(s);
  float ls = lsp[0];
  float v[NCLS];
  float mx = -1e30f;
  for (int k = 0; k < Kc; ++k) {
    float x = ls * simY[(size_t)i * Kc + k] / (hn * tn[k]);
    v[k] = x;
    mx = fmaxf(mx, x);
  }
  float se = 0.f;
  for (int k = 0; k < Kc; ++k) se += expf(v[k] - mx);
  float lse = mx + logf(se);
  for (int k = 0; k < Kc; ++k)
    out[((size_t)b * Kc + k) * Np + n] = v[k] - lse;
}

// ---------------------------------------------------------------------------
// Host orchestration.
// Assumed d_in order: [0]=xyz (8,3,4096), [1]=text_features (512,25), then
// params leaves in JAX pytree order (dict keys sorted):
//   conv1(W,b)=2,3; fp1=4..7; fp2=8..11; fp3=12..15; logit_scale=16;
//   sa1 b0=17..22, b1=23..28, b2=29..34; sa2 b0=35..40, b1=41..46; sa3=47..52.
// Biases are cancelled exactly by the following batchnorm, so only W's and
// logit_scale are consumed.
// ---------------------------------------------------------------------------
struct WPack {
  const _Float16* p;
  int K, N, KC, TNG;  // TNG = padded tile groups (TNpad / GEMM_NT)
};

extern "C" void kernel_launch(void* const* d_in, const int* in_sizes, int n_in,
                              void* d_out, int out_size, void* d_ws, size_t ws_size,
                              hipStream_t stream) {
  (void)in_sizes; (void)n_in; (void)out_size; (void)ws_size;
  auto F = [&](int i) { return (const float*)d_in[i]; };
  const float* xyz_in = F(0);
  const float* text   = F(1);
  const float* lscale = F(16);

  // --- bump allocator over d_ws ---
  char* base = (char*)d_ws;
  size_t off = 0;
  auto alloc_b = [&](size_t bytes) {
    void* p = base + off;
    off += (bytes + 255) & ~(size_t)255;
    return p;
  };
  auto alloc_f = [&](size_t n) { return (float*)alloc_b(n * 4); };
  auto alloc_i = [&](size_t n) { return (int*)alloc_b(n * 4); };

  float* l0xyz = alloc_f((size_t)BB * N0 * 3);
  int*   fps1  = alloc_i((size_t)BB * 512);
  float* nxyz1 = alloc_f((size_t)BB * 512 * 3);
  int*   fps2  = alloc_i((size_t)BB * 128);
  float* nxyz2 = alloc_f((size_t)BB * 128 * 3);
  int*   gidx  = alloc_i((size_t)BB * 512 * 128);
  float* GP    = alloc_f((size_t)BB * 128 * 128 * 323);  // largest grouped input
  float* PING  = alloc_f((size_t)BB * 512 * 128 * 128);  // largest layer output
  float* PONG  = alloc_f((size_t)BB * 512 * 128 * 128);
  float* stats = alloc_f(2048);
  float* l1pts = alloc_f((size_t)BB * 512 * 320);
  float* l2pts = alloc_f((size_t)BB * 128 * 512);
  float* l3pts = alloc_f((size_t)BB * 1024);
  float* fp3o  = alloc_f((size_t)BB * 128 * 256);
  float* fp2o  = alloc_f((size_t)BB * 512 * 128);
  float* fp1o  = alloc_f((size_t)BB * N0 * 128);
  float* tn    = alloc_f(32);

  // --- pack all weights (+ text matrix) into f16 B-fragment layout ---
  const int NW = 26;
  const int widx[NW] = { 17,19,21, 23,25,27, 29,31,33,          // sa1
                         35,37,39, 41,43,45,                    // sa2
                         47,49,51,                              // sa3
                         12,14, 8,10, 4,6,                      // fp3, fp2, fp1
                         2,                                     // conv1
                         1 };                                   // text (512x25)
  const int wK[NW]   = { 6,32,32, 6,64,64, 6,64,96,
                         323,128,128, 323,128,196,
                         515,256,512,
                         1536,256, 576,256, 131,128,
                         128,
                         512 };
  const int wN[NW]   = { 32,32,64, 64,64,128, 64,96,128,
                         128,128,256, 128,196,256,
                         256,512,1024,
                         256,256, 256,128, 128,128,
                         512,
                         25 };
  WPack wp[NW];
  for (int w = 0; w < NW; ++w) {
    int K = wK[w], N = wN[w];
    int KC = (K + 31) / 32;
    int TN = (N + 15) / 16;
    int TNG = (TN + GEMM_NT - 1) / GEMM_NT;
    int TNpad = TNG * GEMM_NT;
    size_t frags = (size_t)TNpad * KC * 32;
    _Float16* p = (_Float16*)alloc_b(frags * 16 * sizeof(_Float16));
    pack_w_kernel<<<dim3((unsigned)((frags + 255) / 256)), dim3(256), 0, stream>>>(
        F(widx[w]), K, N, p, KC, TNpad);
    wp[w] = { p, K, N, KC, TNG };
  }
  auto PW = [&](int idx) -> const WPack& {
    for (int w = 0; w < NW; ++w) if (widx[w] == idx) return wp[w];
    return wp[0];
  };

  auto gemm = [&](const float* A, int M, const WPack& w, float* Y) {
    int tiles = (M / 16) * w.TNG;
    wmma_gemm_kernel<<<dim3(tiles), dim3(32), 0, stream>>>(
        A, w.p, Y, M, w.K, w.N, w.KC, w.TNG);
  };
  auto bn = [&](float* Y, int M, int C, int relu) {
    bn_stats_kernel<<<dim3(C), dim3(256), 0, stream>>>(Y, M, C, stats);
    size_t tot = (size_t)M * C;
    bn_apply_kernel<<<dim3((unsigned)((tot + 255) / 256)), dim3(256), 0, stream>>>(
        Y, stats, tot, C, relu);
  };
  auto layer = [&](const float* X, int M, int widx_, float* Y) {
    const WPack& w = PW(widx_);
    gemm(X, M, w, Y);
    bn(Y, M, w.N, 1);
  };

  // ---- l0: transpose xyz to (B,N,3) ----
  {
    size_t tot = (size_t)BB * N0 * 3;
    xyz_transpose_kernel<<<dim3((unsigned)((tot + 255) / 256)), dim3(256), 0, stream>>>(
        xyz_in, l0xyz);
  }

  // ---- SA1: FPS 4096->512, 3 MSG branches ----
  fps_kernel<<<dim3(BB), dim3(FPS_T), 0, stream>>>(l0xyz, N0, 512, fps1);
  gather_xyz_kernel<<<dim3((BB * 512 + 255) / 256), dim3(256), 0, stream>>>(
      l0xyz, fps1, nxyz1, N0, 512, BB * 512);
  {
    const float rs[3] = {0.1f, 0.2f, 0.4f};
    const int ks[3] = {32, 64, 128};
    const int w0[3] = {17, 23, 29};
    const int d3[3] = {64, 128, 128};
    const int coff[3] = {0, 64, 192};
    for (int br = 0; br < 3; ++br) {
      int Kn = ks[br];
      ball_query_kernel<<<dim3((BB * 512 + 255) / 256), dim3(256), 0, stream>>>(
          l0xyz, nxyz1, N0, 512, Kn, rs[br] * rs[br], gidx, BB * 512);
      size_t rows = (size_t)BB * 512 * Kn;
      group_kernel<<<dim3((unsigned)((rows + 255) / 256)), dim3(256), 0, stream>>>(
          l0xyz, l0xyz, 3, nxyz1, gidx, 512, Kn, N0, GP, rows);
      int M = (int)rows;
      layer(GP,   M, w0[br],     PING);
      layer(PING, M, w0[br] + 2, PONG);
      layer(PONG, M, w0[br] + 4, PING);
      maxpool_kernel<<<dim3((BB * 512 * d3[br] + 255) / 256), dim3(256), 0, stream>>>(
          PING, Kn, d3[br], l1pts, 320, coff[br], BB * 512);
    }
  }

  // ---- SA2: FPS 512->128, 2 MSG branches ----
  fps_kernel<<<dim3(BB), dim3(FPS_T), 0, stream>>>(nxyz1, 512, 128, fps2);
  gather_xyz_kernel<<<dim3((BB * 128 + 255) / 256), dim3(256), 0, stream>>>(
      nxyz1, fps2, nxyz2, 512, 128, BB * 128);
  {
    const float rs[2] = {0.4f, 0.8f};
    const int ks[2] = {64, 128};
    const int w0[2] = {35, 41};
    const int d3[2] = {256, 256};
    const int coff[2] = {0, 256};
    for (int br = 0; br < 2; ++br) {
      int Kn = ks[br];
      ball_query_kernel<<<dim3((BB * 128 + 255) / 256), dim3(256), 0, stream>>>(
          nxyz1, nxyz2, 512, 128, Kn, rs[br] * rs[br], gidx, BB * 128);
      size_t rows = (size_t)BB * 128 * Kn;
      group_kernel<<<dim3((unsigned)((rows + 255) / 256)), dim3(256), 0, stream>>>(
          nxyz1, l1pts, 320, nxyz2, gidx, 128, Kn, 512, GP, rows);
      int M = (int)rows;
      layer(GP,   M, w0[br],     PING);
      layer(PING, M, w0[br] + 2, PONG);
      layer(PONG, M, w0[br] + 4, PING);
      maxpool_kernel<<<dim3((BB * 128 * d3[br] + 255) / 256), dim3(256), 0, stream>>>(
          PING, Kn, d3[br], l2pts, 512, coff[br], BB * 128);
    }
  }

  // ---- SA3: group-all [xyz | l2_pts] -> 515 -> 256 -> 512 -> 1024, max over 128 ----
  {
    int rows = BB * 128;
    concat2_kernel<<<dim3((rows + 255) / 256), dim3(256), 0, stream>>>(
        nxyz2, 3, l2pts, 512, GP, rows, 0);
    layer(GP,   rows, 47, PING);
    layer(PING, rows, 49, PONG);
    layer(PONG, rows, 51, PING);
    maxpool_kernel<<<dim3((BB * 1024 + 255) / 256), dim3(256), 0, stream>>>(
        PING, 128, 1024, l3pts, 1024, 0, BB);
  }

  // ---- FP3: S==1 broadcast, [l2_pts | l3_pts] -> 1536 -> 256 -> 256 ----
  {
    int rows = BB * 128;
    concat2_kernel<<<dim3((rows + 255) / 256), dim3(256), 0, stream>>>(
        l2pts, 512, l3pts, 1024, GP, rows, 128);
    layer(GP,   rows, 12, PING);
    layer(PING, rows, 14, fp3o);
  }

  // ---- FP2: 3-NN interp 128->512, [l1_pts | interp] -> 576 -> 256 -> 128 ----
  {
    int rows = BB * 512;
    fp_interp_kernel<<<dim3((rows + 255) / 256), dim3(256), 0, stream>>>(
        nxyz1, 512, nxyz2, 128, l1pts, 320, fp3o, 256, GP, rows);
    layer(GP,   rows, 8,  PING);
    layer(PING, rows, 10, fp2o);
  }

  // ---- FP1: 3-NN interp 512->4096, [xyz | interp] -> 131 -> 128 -> 128 ----
  {
    int rows = BB * N0;
    fp_interp_kernel<<<dim3((rows + 255) / 256), dim3(256), 0, stream>>>(
        l0xyz, N0, nxyz1, 512, l0xyz, 3, fp2o, 128, GP, rows);
    layer(GP,   rows, 4, PING);
    layer(PING, rows, 6, fp1o);
  }

  // ---- conv1 (BN, no relu) + cosine similarity + log_softmax ----
  {
    int rows = BB * N0;
    gemm(fp1o, rows, PW(2), PING);   // h (pre-norm) lives in PING
    bn(PING, rows, 512, 0);
    tnorm_kernel<<<dim3(1), dim3(32), 0, stream>>>(text, tn);
    gemm(PING, rows, PW(1), PONG);   // h @ T -> PONG (rows x 25)
    final_kernel<<<dim3((rows + 255) / 256), dim3(256), 0, stream>>>(
        PING, PONG, tn, lscale, (float*)d_out, N0, NCLS, 512, rows);
  }
}
